// myGraphAE_16277926051948
// MI455X (gfx1250) — matrix operand order repaired and checked by reading, not compiled, not run on hardware
//
#include <hip/hip_runtime.h>

// ---------------------------------------------------------------------------
// MyGraphAE forward for MI455X (gfx1250).
//   B=1024, N=112. Dominant cost: [1024,7168] x [7168,12544] GEMM (184 GF)
//   on v_wmma_f32_16x16x32_bf16, W_lin streamed via global_load_async_to_lds
//   into double-buffered LDS. bf16 working set (~195 MB) ~ fits 192MB L2.
// Workspace: [ Wb bf16 12544*7168 | act bf16 1024*7168 ]
// ---------------------------------------------------------------------------

typedef __attribute__((ext_vector_type(16))) __bf16 v16bf;
typedef __attribute__((ext_vector_type(8)))  float  v8f;
using u16 = unsigned short;

#define NNODE 112
#define FDIM  7168      // 112*64 flattened features
#define ODIM  12544     // 112*112 outputs per batch

// Single shared-memory window for all kernels; indexed with element offsets so
// every access stays in AS(3) (no generic-pointer null checks, ds_* codegen).
extern __shared__ __align__(16) u16 smemw[];

__device__ __forceinline__ u16 f2bf(float f) {
  unsigned int u = __float_as_uint(f);
  unsigned int r = u + 0x7FFFu + ((u >> 16) & 1u);   // RNE
  return (u16)(r >> 16);
}

__device__ __forceinline__ float mishf(float v) {
  float sp = (v > 20.f) ? v : log1pf(__expf(v));
  return v * tanhf(sp);
}

// A fragment (16x32 bf16 MxK) from LDS, row-major, leading dim ld (elements).
// ISA 7.12.2: lane<16 -> M=lane, K in [0..7] U [16..23]; lane>=16 shifts +8.
__device__ __forceinline__ v16bf ldfragA_s(int off, int ld) {
  const int lane = threadIdx.x & 31;
  const int o = off + (lane & 15) * ld + (lane >> 4) * 8;
  union { v16bf v; uint4 q[2]; } u;
  u.q[0] = *(const uint4*)&smemw[o];
  u.q[1] = *(const uint4*)&smemw[o + 16];
  return u.v;
}

// B fragment (32x16 bf16 KxN) from LDS; source stored transposed Bt[n][k].
// lane<16 -> N=lane, K in [0..7] U [16..23]; lane>=16 shifts +8.
__device__ __forceinline__ v16bf ldfragB_s(int off, int ld) {
  const int lane = threadIdx.x & 31;
  const int o = off + (lane & 15) * ld + (lane >> 4) * 8;
  union { v16bf v; uint4 q[2]; } u;
  u.q[0] = *(const uint4*)&smemw[o];
  u.q[1] = *(const uint4*)&smemw[o + 16];
  return u.v;
}

__device__ __forceinline__ v8f wmma_bf16(v16bf a, v16bf b, v8f c) {
  return __builtin_amdgcn_wmma_f32_16x16x32_bf16(false, a, false, b, (short)0, c,
                                                 false, false);
}

// ---------------------------------------------------------------------------
// Kernel 1: W_lin f32 -> bf16
// ---------------------------------------------------------------------------
__global__ void wconv_kernel(const float* __restrict__ W, u16* __restrict__ Wb,
                             long long total4) {
  long long i = (long long)blockIdx.x * blockDim.x + threadIdx.x;
  long long stride = (long long)gridDim.x * blockDim.x;
  for (; i < total4; i += stride) {
    float4 f = ((const float4*)W)[i];
    unsigned long long pk =
        (unsigned long long)f2bf(f.x) |
        ((unsigned long long)f2bf(f.y) << 16) |
        ((unsigned long long)f2bf(f.z) << 32) |
        ((unsigned long long)f2bf(f.w) << 48);
    *(unsigned long long*)(Wb + 4 * i) = pk;
  }
}

// ---------------------------------------------------------------------------
// Kernel 2: per-batch graph-conv stack (one workgroup = one batch, 4 waves)
// LDS element offsets (u16 units):
// ---------------------------------------------------------------------------
constexpr int ADJ_OFF = 0;                       // [112][128] bf16 adjacency
constexpr int HA_OFF  = ADJ_OFF + NNODE * 128;   // [128][128] h buf A (node-major)
constexpr int HTA_OFF = HA_OFF + 128 * 128;      // [128][128] hT buf A (ch-major)
constexpr int HB_OFF  = HTA_OFF + 128 * 128;     // h buf B
constexpr int HTB_OFF = HB_OFF + 128 * 128;      // hT buf B
constexpr int TM_OFF  = HTB_OFF + 128 * 128;     // [112][128] T = adj@h
constexpr int WT_OFF  = TM_OFF + NNODE * 128;    // [2][128][128] staged weights
constexpr int FP_OFF  = WT_OFF + 2 * 128 * 128;  // f32 tail: dinv[112], lab[112]
constexpr size_t SMEM_GRAPH = (size_t)FP_OFF * 2 + (size_t)(2 * NNODE) * 4;

template <int Cin, int Cout, int Kp, bool LAST>
__device__ __forceinline__ void cheb_layer(int hOff, int hTOff, int oOff,
                                           int oTOff,
                                           const float* __restrict__ Wg,
                                           const float* __restrict__ bg,
                                           u16* __restrict__ act, int b,
                                           int wave, int tid) {
  // Stage weights transposed: Wt[s][cout][cin], cin zero-padded to Kp.
  constexpr int tot = 2 * Cout * Kp;
  for (int idx = tid; idx < tot; idx += 128) {
    int s   = idx / (Cout * Kp);
    int rem = idx - s * (Cout * Kp);
    int co  = rem / Kp;
    int ci  = rem - co * Kp;
    float v = (ci < Cin) ? Wg[(s * Cin + ci) * Cout + co] : 0.f;
    smemw[WT_OFF + s * 16384 + co * 128 + ci] = f2bf(v);
  }
  __syncthreads();

  const int lane = tid & 31;
  const int nl   = lane & 15;
  const int khl  = lane >> 4;

  // Matmul 1: T = adj @ h   (M=112, N=Cin tiles, K=112 padded to 128)
  constexpr int ntT = (Cin + 15) >> 4;
  for (int t = wave; t < 7 * ntT; t += 4) {
    int mt = t % 7, nt = t / 7;
    v8f acc = {0.f, 0.f, 0.f, 0.f, 0.f, 0.f, 0.f, 0.f};
#pragma unroll
    for (int k = 0; k < 128; k += 32) {
      v16bf a  = ldfragA_s(ADJ_OFF + mt * 16 * 128 + k, 128);
      v16bf bb = ldfragB_s(hTOff + nt * 16 * 128 + k, 128);
      acc = wmma_bf16(a, bb, acc);
    }
#pragma unroll
    for (int r = 0; r < 8; ++r) {
      int node = mt * 16 + r + 8 * khl;
      smemw[TM_OFF + node * 128 + nt * 16 + nl] = f2bf(acc[r]);
    }
  }
  __syncthreads();

  // Matmul 2: Y = h@W0 + T@W1 + b ; mish
  constexpr int ntO = Cout >> 4;
  for (int t = wave; t < 7 * ntO; t += 4) {
    int mt = t % 7, nt = t / 7;
    v8f acc = {0.f, 0.f, 0.f, 0.f, 0.f, 0.f, 0.f, 0.f};
#pragma unroll
    for (int k = 0; k < Kp; k += 32) {
      v16bf a0 = ldfragA_s(hOff + mt * 16 * 128 + k, 128);
      v16bf w0 = ldfragB_s(WT_OFF + nt * 16 * 128 + k, 128);
      acc = wmma_bf16(a0, w0, acc);
      v16bf a1 = ldfragA_s(TM_OFF + mt * 16 * 128 + k, 128);
      v16bf w1 = ldfragB_s(WT_OFF + 16384 + nt * 16 * 128 + k, 128);
      acc = wmma_bf16(a1, w1, acc);
    }
#pragma unroll
    for (int r = 0; r < 8; ++r) {
      int node = mt * 16 + r + 8 * khl;
      int c    = nt * 16 + nl;
      float v  = mishf(acc[r] + bg[c]);
      u16 bv   = f2bf(v);
      if (!LAST) {
        smemw[oOff + node * 128 + c] = bv;
        smemw[oTOff + c * 128 + node] = bv;
      } else {
        act[(size_t)b * FDIM + node * 64 + c] = bv;  // final layer, Cout=64
      }
    }
  }
  __syncthreads();
}

__global__ __launch_bounds__(128) void graph_stack_kernel(
    const float* __restrict__ x, const float* __restrict__ edge,
    const int* __restrict__ label, const float* __restrict__ emb,
    const float* __restrict__ We1, const float* __restrict__ be1,
    const float* __restrict__ We2, const float* __restrict__ be2,
    const float* __restrict__ Wg1, const float* __restrict__ bg1,
    const float* __restrict__ Wg2, const float* __restrict__ bg2,
    u16* __restrict__ act) {
  float* adjF = (float*)&smemw[WT_OFF];   // [112][116] f32 (temp, pre-weights)
  float* dinv = (float*)&smemw[FP_OFF];   // [112]
  float* labF = dinv + NNODE;             // [112]

  const int tid  = threadIdx.x;
  const int wave = tid >> 5;
  const int b    = blockIdx.x;

  // Zero padded regions (K/row padding relies on these zeros).
  for (int i = tid; i < 128 * 128; i += 128) {
    smemw[HA_OFF + i] = 0; smemw[HTA_OFF + i] = 0;
    smemw[HB_OFF + i] = 0; smemw[HTB_OFF + i] = 0;
  }
  for (int i = tid; i < NNODE * 128; i += 128) smemw[TM_OFF + i] = 0;

  // Stage edge matrix + label embedding.
  const float* E = edge + (size_t)b * (NNODE * NNODE);
  for (int idx = tid; idx < NNODE * NNODE; idx += 128) {
    int i = idx / NNODE, j = idx - i * NNODE;
    adjF[i * 116 + j] = E[idx];
  }
  int lb = label[b];
  for (int i = tid; i < NNODE; i += 128) labF[i] = emb[lb * NNODE + i];
  __syncthreads();

  // Degree + D^{-1/2}
  if (tid < NNODE) {
    float s = 0.f;
    const float* row = adjF + tid * 116;
    for (int j = 0; j < NNODE; ++j) s += row[j];
    dinv[tid] = (s > 0.f) ? rsqrtf(fmaxf(s, 1e-12f)) : 0.f;
  }
  __syncthreads();

  // Normalized adjacency in bf16, K padded to 128 with zeros.
  for (int idx = tid; idx < NNODE * 128; idx += 128) {
    int i = idx >> 7, j = idx & 127;
    float v = (j < NNODE) ? dinv[i] * adjF[i * 116 + j] * dinv[j] : 0.f;
    smemw[ADJ_OFF + idx] = f2bf(v);
  }

  // h0 = [x | lab]  (Cin=4, rest zero-padded)
  const float* X = x + (size_t)b * NNODE * 3;
  for (int i = tid; i < NNODE; i += 128) {
    float c0 = X[i * 3 + 0], c1 = X[i * 3 + 1], c2 = X[i * 3 + 2], l0 = labF[i];
    smemw[HA_OFF + i * 128 + 0] = f2bf(c0);
    smemw[HA_OFF + i * 128 + 1] = f2bf(c1);
    smemw[HA_OFF + i * 128 + 2] = f2bf(c2);
    smemw[HA_OFF + i * 128 + 3] = f2bf(l0);
    smemw[HTA_OFF + 0 * 128 + i] = f2bf(c0);
    smemw[HTA_OFF + 1 * 128 + i] = f2bf(c1);
    smemw[HTA_OFF + 2 * 128 + i] = f2bf(c2);
    smemw[HTA_OFF + 3 * 128 + i] = f2bf(l0);
  }
  __syncthreads();

  cheb_layer< 4,  64, 32, false>(HA_OFF, HTA_OFF, HB_OFF, HTB_OFF, We1, be1, act, b, wave, tid);
  cheb_layer<64,  64, 64, false>(HB_OFF, HTB_OFF, HA_OFF, HTA_OFF, We2, be2, act, b, wave, tid);
  // concat label column (channel 64); channels 65..95 are still zero.
  for (int i = tid; i < NNODE; i += 128) {
    u16 lv = f2bf(labF[i]);
    smemw[HA_OFF + i * 128 + 64] = lv;
    smemw[HTA_OFF + 64 * 128 + i] = lv;
  }
  __syncthreads();
  cheb_layer< 65, 128,  96, false>(HA_OFF, HTA_OFF, HB_OFF, HTB_OFF, Wg1, bg1, act, b, wave, tid);
  cheb_layer<128,  64, 128, true >(HB_OFF, HTB_OFF, 0, 0, Wg2, bg2, act, b, wave, tid);
}

// ---------------------------------------------------------------------------
// Kernel 3: out = tanh(act @ Wb^T + b_lin)   [1024,7168] x [12544,7168]^T
// Block tile 128x128, 8 waves x (4M x 2N) WMMA tiles, KC=64, LDS double-buffer
// filled with global_load_async_to_lds_b128 (ASYNCcnt-pipelined).
// LDS: buf{0,1} x (A[128][64] | B[128][64]) bf16 = 64 KB.
// ---------------------------------------------------------------------------
constexpr int GEMM_BUF_E = 128 * 64;                 // 8192 elements per tile
constexpr size_t SMEM_GEMM = 2 * 2 * GEMM_BUF_E * sizeof(u16);  // 65536 B

__global__ __launch_bounds__(256) void lin_gemm_kernel(
    const u16* __restrict__ A,    // [1024][7168] bf16
    const u16* __restrict__ Bw,   // [12544][7168] bf16 (row n contiguous in k)
    const float* __restrict__ bias, float* __restrict__ out) {
  const int tid  = threadIdx.x;
  const int lane = tid & 31;
  const int wave = tid >> 5;
  const int nBase = blockIdx.x * 128;
  const int mBase = blockIdx.y * 128;
  const int wm = wave & 1;    // 2 wave-rows, each 4 M-tiles
  const int wn = wave >> 1;   // 4 wave-cols, each 2 N-tiles

  // Issue one K-chunk (64) of A+B into LDS buffer `sel` via async DMA.
  auto issue = [&](int kc, int sel) {
    const int base = sel * (2 * GEMM_BUF_E);
#pragma unroll
    for (int i = 0; i < 4; ++i) {
      int vi  = tid + 256 * i;
      int row = vi >> 3;
      int k8  = (vi & 7) * 8;
      unsigned ldsA = (unsigned)(unsigned long long)(const void*)
                          &smemw[base + row * 64 + k8];
      unsigned long long gA = (unsigned long long)(const void*)
                          (A + (size_t)(mBase + row) * FDIM + kc + k8);
      asm volatile("global_load_async_to_lds_b128 %0, %1, off"
                   :: "v"(ldsA), "v"(gA) : "memory");
      unsigned ldsB = (unsigned)(unsigned long long)(const void*)
                          &smemw[base + GEMM_BUF_E + row * 64 + k8];
      unsigned long long gB = (unsigned long long)(const void*)
                          (Bw + (size_t)(nBase + row) * FDIM + kc + k8);
      asm volatile("global_load_async_to_lds_b128 %0, %1, off"
                   :: "v"(ldsB), "v"(gB) : "memory");
    }
  };

  v8f acc[4][2];
#pragma unroll
  for (int i = 0; i < 4; ++i)
#pragma unroll
    for (int j = 0; j < 2; ++j)
      acc[i][j] = (v8f){0.f, 0.f, 0.f, 0.f, 0.f, 0.f, 0.f, 0.f};

  issue(0, 0);
  int sel = 0;
  for (int kc = 0; kc < FDIM; kc += 64) {
    const bool more = (kc + 64) < FDIM;
    if (more) {
      issue(kc + 64, sel ^ 1);
      // 8 ops just issued; async loads complete in order, so <=8 outstanding
      // means the current buffer's chunk has fully landed.
      asm volatile("s_wait_asynccnt 0x8" ::: "memory");
    } else {
      asm volatile("s_wait_asynccnt 0x0" ::: "memory");
    }
    __syncthreads();

    const int ab = sel * (2 * GEMM_BUF_E);
    const int bb = ab + GEMM_BUF_E;
#pragma unroll
    for (int ks = 0; ks < 64; ks += 32) {
      v16bf af[4], bfr[2];
#pragma unroll
      for (int i = 0; i < 4; ++i)
        af[i] = ldfragA_s(ab + (wm * 64 + i * 16) * 64 + ks, 64);
#pragma unroll
      for (int j = 0; j < 2; ++j)
        bfr[j] = ldfragB_s(bb + (wn * 32 + j * 16) * 64 + ks, 64);
#pragma unroll
      for (int i = 0; i < 4; ++i)
#pragma unroll
        for (int j = 0; j < 2; ++j)
          acc[i][j] = wmma_bf16(af[i], bfr[j], acc[i][j]);
    }
    __syncthreads();   // everyone done reading `sel` before it is refilled
    sel ^= 1;
  }

  // Epilogue: bias + tanh, direct store.
  const int nl  = lane & 15;
  const int khl = lane >> 4;
#pragma unroll
  for (int i = 0; i < 4; ++i)
#pragma unroll
    for (int j = 0; j < 2; ++j) {
      int n = nBase + wn * 32 + j * 16 + nl;
      float bn = bias[n];
#pragma unroll
      for (int r = 0; r < 8; ++r) {
        int m = mBase + wm * 64 + i * 16 + r + 8 * khl;
        out[(size_t)m * ODIM + n] = tanhf(acc[i][j][r] + bn);
      }
    }
}

// ---------------------------------------------------------------------------
// Kernel 4: in-place symmetrize + arctanh(0.999*) + unit diagonal.
// Thread owns pair (i<=j): reads both, writes both -> race free.
// ---------------------------------------------------------------------------
__global__ void sym_kernel(float* __restrict__ out) {
  int p = blockIdx.x * blockDim.x + threadIdx.x;   // 0..12543
  if (p >= ODIM) return;
  int b = blockIdx.y;
  int i = p / NNODE, j = p - i * NNODE;
  if (j < i) return;
  float* O = out + (size_t)b * ODIM;
  if (i == j) { O[p] = 1.0f; return; }
  float a = O[i * NNODE + j];
  float c = O[j * NNODE + i];
  float r = atanhf(0.999f * 0.5f * (a + c));
  O[i * NNODE + j] = r;
  O[j * NNODE + i] = r;
}

// ---------------------------------------------------------------------------
extern "C" void kernel_launch(void* const* d_in, const int* in_sizes, int n_in,
                              void* d_out, int out_size, void* d_ws, size_t ws_size,
                              hipStream_t stream) {
  const float* x    = (const float*)d_in[0];
  const float* edge = (const float*)d_in[1];
  const int*   lab  = (const int*)d_in[2];
  const float* We1  = (const float*)d_in[3];
  const float* be1  = (const float*)d_in[4];
  const float* We2  = (const float*)d_in[5];
  const float* be2  = (const float*)d_in[6];
  const float* Wg1  = (const float*)d_in[7];
  const float* bg1  = (const float*)d_in[8];
  const float* Wg2  = (const float*)d_in[9];
  const float* bg2  = (const float*)d_in[10];
  const float* Wlin = (const float*)d_in[11];
  const float* blin = (const float*)d_in[12];
  const float* emb  = (const float*)d_in[13];
  float* out = (float*)d_out;

  u16* Wb  = (u16*)d_ws;                       // 12544*7168 bf16 (~180 MB)
  u16* act = Wb + (size_t)ODIM * FDIM;         // 1024*7168 bf16 (~14.7 MB)

  // Allow >64KB dynamic LDS (WGP has 320KB).
  (void)hipFuncSetAttribute((const void*)graph_stack_kernel,
                            hipFuncAttributeMaxDynamicSharedMemorySize,
                            (int)SMEM_GRAPH);
  (void)hipFuncSetAttribute((const void*)lin_gemm_kernel,
                            hipFuncAttributeMaxDynamicSharedMemorySize,
                            (int)SMEM_GEMM);

  // 1) W_lin -> bf16
  wconv_kernel<<<4096, 256, 0, stream>>>(Wlin, Wb,
                                         (long long)((size_t)ODIM * FDIM / 4));
  // 2) graph conv stack -> bf16 activations
  graph_stack_kernel<<<1024, 128, SMEM_GRAPH, stream>>>(
      x, edge, lab, emb, We1, be1, We2, be2, Wg1, bg1, Wg2, bg2, act);
  // 3) big GEMM + bias + tanh  (98 x 8 blocks of 128x128)
  lin_gemm_kernel<<<dim3(ODIM / 128, 1024 / 128), 256, SMEM_GEMM, stream>>>(
      act, Wb, blin, out);
  // 4) symmetrize + arctanh + diag
  sym_kernel<<<dim3(ODIM / 256, 1024), 256, 0, stream>>>(out);

  (void)in_sizes; (void)n_in; (void)out_size; (void)ws_size;
}
